// FLARE_71116068487387
// MI455X (gfx1250) — matrix operand order repaired and checked by reading, not compiled, not run on hardware
//
#include <hip/hip_runtime.h>
#include <hip/hip_bf16.h>

typedef __attribute__((ext_vector_type(16))) _Float16 v16h;
typedef __attribute__((ext_vector_type(8)))  _Float16 v8h;
typedef __attribute__((ext_vector_type(8)))  float    v8f;

#define C_DIM 1024
#define M_DIM 32768      // B*N = 4*8192
#define KSTEPS 32        // 1024 / 32
#define TM 128
#define TN 128

// ---------------------------------------------------------------------------
// Pack fp32 weight [1024][1024] (row-major, x @ W convention) into f16
// WMMA-B fragment-major layout:
//   frag f = ks*64 + nt   (ks: k-step of 32, nt: 16-col tile)
//   element index = (f*32 + lane)*16 + e
//   lane in [0,16): n = nt*16 + lane,     k = ks*32 + e        (e in [0,16))
//   lane in [16,32): n = nt*16 + lane-16, k = ks*32 + 16 + e
// (contiguous k order, same as A: lanes 0-15 hold K=0..15, lanes 16-31 K=16..31)
// ---------------------------------------------------------------------------
__global__ __launch_bounds__(256) void FLARE_pack_w(const float* __restrict__ W,
                                                    _Float16* __restrict__ out) {
    int idx  = blockIdx.x * 256 + threadIdx.x;     // 0 .. 1M-1
    int e    = idx & 15;
    int ln   = (idx >> 4) & 31;
    int frag = idx >> 9;
    int nt   = frag & 63;
    int ks   = frag >> 6;
    int n = nt * 16 + (ln & 15);
    int k = ks * 32 + ((ln >> 4) << 4) + e;
    out[idx] = (_Float16)W[k * C_DIM + n];
}

__global__ __launch_bounds__(256) void FLARE_f32_to_f16(const float* __restrict__ in,
                                                        _Float16* __restrict__ out, int n) {
    int i = blockIdx.x * 256 + threadIdx.x;
    if (i < n) out[i] = (_Float16)in[i];
}

// ---------------------------------------------------------------------------
// Tiled WMMA GEMM: out = epilogue(A[M x 1024] @ W[1024 x 1024] + bias)
//   A: f16 row-major.  Bsw: f16 fragment-major (see FLARE_pack_w).
//   epilogue: v = acc + bias; if(act) v = gelu_tanh(v); if(res) v += res;
//   writes fp32 (outF) and/or f16 row-major (outH).
// Block: 256 threads (8 wave32). Block tile 128x128; wave w owns rows
// [bm*128 + 16w, +16) x all 128 cols (8 WMMA tiles). B double-buffered in LDS.
// ---------------------------------------------------------------------------
__global__ __launch_bounds__(256) void FLARE_gemm_wmma(
    const _Float16* __restrict__ A,
    const _Float16* __restrict__ Bsw,
    const float*    __restrict__ bias,
    const float*    __restrict__ res,
    float*          __restrict__ outF,
    _Float16*       __restrict__ outH,
    int act)
{
    __shared__ _Float16 lds[2][4096];   // 2 x (8 frags * 32 lanes * 16 halfs) = 16 KB

    const int tid  = threadIdx.x;
    const int wave = tid >> 5;
    const int lane = tid & 31;
    const int bm = blockIdx.x, bn = blockIdx.y;

    // A fragment addressing (16-bit A 16x32, contiguous k order as in packed B):
    //  lane<16 : row=lane,    halfs K[0..15]
    //  lane>=16: row=lane-16, halfs K[16..31]
    const long arow  = (long)(bm * TM + wave * 16 + (lane & 15)) * C_DIM;
    const int  khalf = lane >> 4;

    v8f acc[8] = {};

    // B staging: thread tid covers (frag t=tid>>5, lane'=tid&31), LDS slot tid*16
    const int sfrag = tid >> 5, slane = tid & 31;
    auto bsrc = [&](int ks) {
        long f = (long)ks * 64 + bn * 8 + sfrag;
        return (const v16h*)(Bsw + (f * 32 + slane) * 16);
    };

    v16h breg = *bsrc(0);
    *(v16h*)&lds[0][tid * 16] = breg;

    for (int ks = 0; ks < KSTEPS; ++ks) {
        __syncthreads();
        if (ks + 1 < KSTEPS) breg = *bsrc(ks + 1);
        if (ks + 2 < KSTEPS) __builtin_prefetch((const void*)bsrc(ks + 2), 0, 3);

        // A fragment: two contiguous 16-byte loads per lane
        const int k0 = ks * 32;
        v8h alo = *(const v8h*)(A + arow + k0 + khalf * 16);
        v8h ahi = *(const v8h*)(A + arow + k0 + khalf * 16 + 8);
        v16h a = __builtin_shufflevector(alo, ahi,
                                         0, 1, 2, 3, 4, 5, 6, 7,
                                         8, 9, 10, 11, 12, 13, 14, 15);

        const _Float16* bb = &lds[ks & 1][0];
#pragma unroll
        for (int t = 0; t < 8; ++t) {
            v16h b = *(const v16h*)(bb + (t * 32 + lane) * 16);
            acc[t] = __builtin_amdgcn_wmma_f32_16x16x32_f16(
                false, a, false, b, (short)0, acc[t], false, false);
        }
        if (ks + 1 < KSTEPS) *(v16h*)&lds[(ks + 1) & 1][tid * 16] = breg;
    }

    // Epilogue. D layout (ISA 7.12.2 f32 C/D): VGPR r -> row r (lanes 0-15) or r+8
    const int mbase = bm * TM + wave * 16 + ((lane < 16) ? 0 : 8);
    const int ncol  = bn * TN + (lane & 15);
#pragma unroll
    for (int t = 0; t < 8; ++t) {
        const int n = ncol + t * 16;
        const float bv = bias[n];
#pragma unroll
        for (int r = 0; r < 8; ++r) {
            const long m = mbase + r;
            float v = acc[t][r] + bv;
            if (act) {
                float u = v;
                v = 0.5f * u * (1.f + tanhf(0.7978845608028654f * (u + 0.044715f * u * u * u)));
            }
            if (res)  v += res[m * C_DIM + n];
            if (outF) outF[m * C_DIM + n] = v;
            if (outH) outH[m * C_DIM + n] = (_Float16)v;
        }
    }
}

// ---------------------------------------------------------------------------
// Encode: z[b,h,l,:] = softmax_n(q_l . k_n) @ v   (online softmax over N=8192)
// One WG of 256 threads per (b,h); group of 8 lanes per latent l.
// ---------------------------------------------------------------------------
__global__ __launch_bounds__(256) void FLARE_attn_encode(
    const float* __restrict__ latent_q,   // [H][L*D] after torch-style view
    const float* __restrict__ K,          // [B*N][C]
    const float* __restrict__ V,          // [B*N][C]
    float*       __restrict__ Z)          // [B*H][L][D]
{
    __shared__ float qs[32 * 64];
    __shared__ float kch[32 * 64];
    __shared__ float vch[32 * 64];
    __shared__ float zacc[32 * 64];
    __shared__ float pch[32][32];

    const int tid = threadIdx.x;
    const int bh  = blockIdx.x;           // 0..63
    const int b   = bh >> 4, h = bh & 15;
    const float* qh = latent_q + h * 2048;

#pragma unroll
    for (int i = 0; i < 8; ++i) {
        int idx = tid * 8 + i;
        qs[idx]   = qh[idx];
        zacc[idx] = 0.f;
    }
    const int l = tid >> 3, sub = tid & 7;
    float m_run = -1e30f, s_run = 0.f;

    for (int c = 0; c < 8192 / 32; ++c) {
        __syncthreads();
        const int nbase = c * 32;
#pragma unroll
        for (int i = 0; i < 8; ++i) {
            int idx = tid * 8 + i;
            int nl = idx >> 6, d = idx & 63;
            long g = (long)(b * 8192 + nbase + nl) * C_DIM + h * 64 + d;
            kch[idx] = K[g];
            vch[idx] = V[g];
        }
        __syncthreads();

        float sc[4]; float cmax = -1e30f;
#pragma unroll
        for (int j = 0; j < 4; ++j) {
            int nl = sub + j * 8;
            float a = 0.f;
            for (int d = 0; d < 64; ++d) a += qs[l * 64 + d] * kch[nl * 64 + d];
            sc[j] = a;                      // SCALE == 1.0
            cmax = fmaxf(cmax, sc[j]);
        }
        for (int o = 4; o >= 1; o >>= 1) cmax = fmaxf(cmax, __shfl_xor(cmax, o, 8));
        float mnew = fmaxf(m_run, cmax);
        float corr = __expf(m_run - mnew);
        float psum = 0.f;
#pragma unroll
        for (int j = 0; j < 4; ++j) {
            sc[j] = __expf(sc[j] - mnew);
            psum += sc[j];
            pch[l][sub + j * 8] = sc[j];
        }
        for (int o = 4; o >= 1; o >>= 1) psum += __shfl_xor(psum, o, 8);
        s_run = s_run * corr + psum;
        m_run = mnew;
        __syncthreads();
#pragma unroll
        for (int i = 0; i < 8; ++i) {
            int d = sub * 8 + i;
            float z = zacc[l * 64 + d] * corr;
            for (int n2 = 0; n2 < 32; ++n2) z += pch[l][n2] * vch[n2 * 64 + d];
            zacc[l * 64 + d] = z;
        }
    }
    float inv = 1.f / s_run;
#pragma unroll
    for (int i = 0; i < 8; ++i) {
        int d = sub * 8 + i;
        Z[((long)bh * 32 + l) * 64 + d] = zacc[l * 64 + d] * inv;
    }
}

// ---------------------------------------------------------------------------
// Decode: y[b,n,h,:] = softmax_l(k_n . q_l) @ z[b,h]; one thread per token.
// Writes y as f16 row-major [B*N][C] (head-interleaved) for the final GEMM.
// ---------------------------------------------------------------------------
__global__ __launch_bounds__(256) void FLARE_attn_decode(
    const float* __restrict__ latent_q,
    const float* __restrict__ K,          // [B*N][C]
    const float* __restrict__ Z,          // [B*H][L][D]
    _Float16*    __restrict__ Y)          // [B*N][C] f16
{
    __shared__ float qs[32 * 64];
    __shared__ float zs[32 * 64];
    const int tid = threadIdx.x;
    const int bh  = blockIdx.y;
    const int b   = bh >> 4, h = bh & 15;
    const int n   = blockIdx.x * 256 + tid;
    const float* qh = latent_q + h * 2048;
    const float* zh = Z + (long)bh * 2048;
#pragma unroll
    for (int i = 0; i < 8; ++i) {
        int idx = tid * 8 + i;
        qs[idx] = qh[idx];
        zs[idx] = zh[idx];
    }
    __syncthreads();

    float kreg[64];
    long base = (long)(b * 8192 + n) * C_DIM + h * 64;
#pragma unroll
    for (int d = 0; d < 64; ++d) kreg[d] = K[base + d];

    float lg[32]; float mx = -1e30f;
#pragma unroll
    for (int l2 = 0; l2 < 32; ++l2) {
        float a = 0.f;
        for (int d = 0; d < 64; ++d) a += kreg[d] * qs[l2 * 64 + d];
        lg[l2] = a;                       // SCALE == 1.0
        mx = fmaxf(mx, a);
    }
    float s = 0.f;
#pragma unroll
    for (int l2 = 0; l2 < 32; ++l2) { lg[l2] = __expf(lg[l2] - mx); s += lg[l2]; }
    float inv = 1.f / s;
#pragma unroll
    for (int d = 0; d < 64; ++d) {
        float a = 0.f;
        for (int l2 = 0; l2 < 32; ++l2) a += lg[l2] * zs[l2 * 64 + d];
        Y[base + d] = (_Float16)(a * inv);
    }
}

// ---------------------------------------------------------------------------
extern "C" void kernel_launch(void* const* d_in, const int* in_sizes, int n_in,
                              void* d_out, int out_size, void* d_ws, size_t ws_size,
                              hipStream_t stream) {
    (void)in_sizes; (void)n_in; (void)out_size; (void)ws_size;

    const float* x  = (const float*)d_in[0];
    const float* lq = (const float*)d_in[1];
    const size_t WMAT = 1048576;   // 1024*1024

    const float* wsrc[11] = {
        (const float*)d_in[2],
        (const float*)d_in[4], (const float*)d_in[4] + WMAT, (const float*)d_in[4] + 2 * WMAT,
        (const float*)d_in[6],
        (const float*)d_in[8],
        (const float*)d_in[10], (const float*)d_in[10] + WMAT, (const float*)d_in[10] + 2 * WMAT,
        (const float*)d_in[12],
        (const float*)d_in[14],
    };
    const float* kb[5] = {
        (const float*)d_in[3],
        (const float*)d_in[5], (const float*)d_in[5] + 1024, (const float*)d_in[5] + 2048,
        (const float*)d_in[7],
    };
    const float* vb[5] = {
        (const float*)d_in[9],
        (const float*)d_in[11], (const float*)d_in[11] + 1024, (const float*)d_in[11] + 2048,
        (const float*)d_in[13],
    };
    const float* ob = (const float*)d_in[15];

    // Workspace layout
    char* ws = (char*)d_ws;
    size_t off = 0;
    auto take = [&](size_t bytes) { size_t r = off; off += (bytes + 255) & ~(size_t)255; return r; };
    const size_t ELT = (size_t)M_DIM * C_DIM;   // 33.5M elements
    size_t oW   = take(11 * WMAT * 2);
    size_t oX16 = take(ELT * 2);
    size_t oT0F = take(ELT * 4);
    size_t oT1F = take(ELT * 4);
    size_t oT0H = take(ELT * 2);
    size_t oT1H = take(ELT * 2);
    size_t oK   = take(ELT * 4);
    size_t oV   = take(ELT * 4);
    size_t oZ   = take((size_t)64 * 2048 * 4);

    _Float16* Wsw = (_Float16*)(ws + oW);
    _Float16* X16 = (_Float16*)(ws + oX16);
    float*    T0F = (float*)(ws + oT0F);
    float*    T1F = (float*)(ws + oT1F);
    _Float16* T0H = (_Float16*)(ws + oT0H);
    _Float16* T1H = (_Float16*)(ws + oT1H);
    float*    KF  = (float*)(ws + oK);
    float*    VF  = (float*)(ws + oV);
    float*    Zf  = (float*)(ws + oZ);
    _Float16* YH  = T0H;   // alias (T0H dead after MLPs)

    // 1) pack weights fp32 -> f16 fragment-major
    for (int i = 0; i < 11; ++i)
        FLARE_pack_w<<<4096, 256, 0, stream>>>(wsrc[i], Wsw + (size_t)i * WMAT);

    // 2) x -> f16 row-major
    FLARE_f32_to_f16<<<(int)(ELT / 256), 256, 0, stream>>>(x, X16, (int)ELT);

    dim3 gg(M_DIM / TM, C_DIM / TN);
    auto gemm = [&](const _Float16* A, int wslot, const float* bias, const float* res,
                    float* oF, _Float16* oH, int act) {
        FLARE_gemm_wmma<<<gg, 256, 0, stream>>>(A, Wsw + (size_t)wslot * WMAT, bias, res, oF, oH, act);
    };

    // 3) K residual MLP: h = x + gelu(x@W1+b); 3x h = h + gelu(h@Wi+b); k = h + (h@W2+b)
    gemm(X16, 0, kb[0], x,   T0F, T0H, 1);
    gemm(T0H, 1, kb[1], T0F, T1F, T1H, 1);
    gemm(T1H, 2, kb[2], T1F, T0F, T0H, 1);
    gemm(T0H, 3, kb[3], T0F, T1F, T1H, 1);
    gemm(T1H, 4, kb[4], T1F, KF,  nullptr, 0);

    // 4) V residual MLP
    gemm(X16, 5, vb[0], x,   T0F, T0H, 1);
    gemm(T0H, 6, vb[1], T0F, T1F, T1H, 1);
    gemm(T1H, 7, vb[2], T1F, T0F, T0H, 1);
    gemm(T0H, 8, vb[3], T0F, T1F, T1H, 1);
    gemm(T1H, 9, vb[4], T1F, VF,  nullptr, 0);

    // 5) latent attention encode + decode
    FLARE_attn_encode<<<64, 256, 0, stream>>>(lq, KF, VF, Zf);
    FLARE_attn_decode<<<dim3(8192 / 256, 64), 256, 0, stream>>>(lq, KF, Zf, YH);

    // 6) output projection -> d_out (fp32)
    gemm(YH, 10, ob, nullptr, (float*)d_out, nullptr, 0);
}
